// KAGAT_6640019439797
// MI455X (gfx1250) — matrix operand order repaired and checked by reading, not compiled, not run on hardware
//
#include <hip/hip_runtime.h>
#include <hip/hip_bf16.h>

// ---------------- problem constants (match reference) ----------------
#define NN    50000
#define EE    800000
#define ETOT  (EE + NN)      // edges + self loops
#define NGG   64
#define DD    128
#define HH    4
#define LL    3
#define NBB   7
#define NCC   10
#define KPACK 1024           // 128 channels * (1 silu + 7 bases)
#define MT    32             // nodes per GEMM tile (2 x 16 M-subtiles)

typedef __bf16 v16bf __attribute__((ext_vector_type(16)));
typedef __bf16 v8bf  __attribute__((ext_vector_type(8)));
typedef float  v8f   __attribute__((ext_vector_type(8)));

static __device__ __forceinline__ unsigned short f2bf(float f) {
  unsigned u = __float_as_uint(f);
  u += 0x7FFFu + ((u >> 16) & 1u);          // round-to-nearest-even
  return (unsigned short)(u >> 16);
}
static __device__ __forceinline__ float siluf(float x) { return x / (1.f + __expf(-x)); }

// Cox-de Boor, uniform grid g[j] = 0.5*j - 2.5 (GS=4, K=3) -> 7 bases
static __device__ __forceinline__ void bspline7(float x, float* o) {
  float b[10];
#pragma unroll
  for (int j = 0; j < 10; ++j) {
    float g0 = 0.5f * j - 2.5f;
    b[j] = (x >= g0 && x < g0 + 0.5f) ? 1.f : 0.f;
  }
#pragma unroll
  for (int p = 1; p <= 3; ++p) {
    float inv = 1.f / (0.5f * (float)p);
#pragma unroll
    for (int j = 0; j < 10 - p; ++j) {
      float gj = 0.5f * j - 2.5f;
      float gr = gj + 0.5f * (float)(p + 1);
      b[j] = (x - gj) * inv * b[j] + (gr - x) * inv * b[j + 1];
    }
  }
#pragma unroll
  for (int j = 0; j < 7; ++j) o[j] = b[j];
}

// order-preserving float<->uint key for atomicMax-based segment max
static __device__ __forceinline__ unsigned fkey(float f) {
  unsigned u = __float_as_uint(f);
  return (u >> 31) ? ~u : (u | 0x80000000u);
}
static __device__ __forceinline__ float funkey(unsigned k) {
  unsigned u = (k & 0x80000000u) ? (k & 0x7FFFFFFFu) : ~k;
  return __uint_as_float(u);
}

// ---------------- utility kernels ----------------
__global__ void zero_b32(unsigned* __restrict__ p, long n) {
  long t = (long)blockIdx.x * blockDim.x + threadIdx.x;
  if (t < n) p[t] = 0u;
}

// pack [base_w | spline_w*scaler] -> bf16 [L][128 out][1024 k]
__global__ void pack_weights(const float* __restrict__ base_w,
                             const float* __restrict__ spline_w,
                             const float* __restrict__ scaler,
                             unsigned short* __restrict__ wpack) {
  int t = blockIdx.x * blockDim.x + threadIdx.x;   // over L*128*128
  if (t >= LL * DD * DD) return;
  int i = t % DD;                                  // input channel
  float sc = scaler[t];
  unsigned short* dstp = wpack + (size_t)(t / DD) * KPACK + (size_t)i * 8;
  dstp[0] = f2bf(base_w[t]);
  const float* sw = spline_w + (size_t)t * NBB;
#pragma unroll
  for (int b = 0; b < NBB; ++b) dstp[1 + b] = f2bf(sw[b] * sc);
}

// ---------------- fused KANLinear GEMM: [32 nodes x 1024] x [1024 x 128] ----------------
// 8 waves; wave w owns output cols [16w, 16w+16) for both 16-row M-subtiles.
// Each B fragment (global, L2-resident) feeds two WMMAs.
__global__ __launch_bounds__(256) void kan_gemm(const float* __restrict__ xin,
                                                const unsigned short* __restrict__ wp,
                                                float* __restrict__ hlin, int nNodes) {
  __shared__ unsigned short As[MT * KPACK];        // 64 KB bf16 A-tile
  const int node0 = blockIdx.x * MT;
  const int tid = threadIdx.x;
  const bool full = (node0 + MT) <= nNodes;

  // build A tile: silu + 7 spline bases per (node, channel)
#pragma unroll
  for (int it = 0; it < (MT * DD) / 256; ++it) {
    int idx = tid + it * 256;                      // 0..MT*128-1
    int m = idx >> 7, ch = idx & 127;
    int node = node0 + m;
    union { unsigned short u[8]; uint4 v; } pk;
    if (full || node < nNodes) {
      float x = xin[(size_t)node * DD + ch];
      float bs[7]; bspline7(x, bs);
      pk.u[0] = f2bf(siluf(x));
#pragma unroll
      for (int b = 0; b < 7; ++b) pk.u[1 + b] = f2bf(bs[b]);
    } else {
      pk.v = make_uint4(0u, 0u, 0u, 0u);
    }
    *(uint4*)&As[(size_t)m * KPACK + (size_t)ch * 8] = pk.v;
  }
  __syncthreads();

  const int lane = tid & 31;
  const int wave = tid >> 5;                       // N-subtile 0..7
  const int nIdx = lane & 15;
  const int half = lane >> 4;
  const int outCol = wave * 16 + nIdx;
  const unsigned short* wrow = wp + (size_t)outCol * KPACK;
  const unsigned short* arow0 = &As[(size_t)nIdx * KPACK];          // rows 0..15
  const unsigned short* arow1 = &As[(size_t)(16 + nIdx) * KPACK];   // rows 16..31

  v8f acc0 = {0.f, 0.f, 0.f, 0.f, 0.f, 0.f, 0.f, 0.f};
  v8f acc1 = {0.f, 0.f, 0.f, 0.f, 0.f, 0.f, 0.f, 0.f};
#pragma unroll 4
  for (int k = 0; k < KPACK; k += 32) {
    // B frag (16-bit 32x16): per half-wave K k+half*8..+7 then k+16+half*8..+7
    v8bf b0 = *(const v8bf*)&wrow[k + half * 8];
    v8bf b1 = *(const v8bf*)&wrow[k + 16 + half * 8];
    v16bf Bf;
#pragma unroll
    for (int t2 = 0; t2 < 8; ++t2) { Bf[t2] = b0[t2]; Bf[t2 + 8] = b1[t2]; }
    // A frags (16-bit 16x32 layout): per half-wave K k+half*8..+7 then k+16+half*8..+7
    v8bf a0 = *(const v8bf*)&arow0[k + half * 8];
    v8bf a1 = *(const v8bf*)&arow0[k + 16 + half * 8];
    v8bf c0 = *(const v8bf*)&arow1[k + half * 8];
    v8bf c1 = *(const v8bf*)&arow1[k + 16 + half * 8];
    v16bf Af, Cf;
#pragma unroll
    for (int t2 = 0; t2 < 8; ++t2) {
      Af[t2] = a0[t2]; Af[t2 + 8] = a1[t2];
      Cf[t2] = c0[t2]; Cf[t2 + 8] = c1[t2];
    }
    acc0 = __builtin_amdgcn_wmma_f32_16x16x32_bf16(false, Af, false, Bf,
                                                   (short)0, acc0, false, false);
    acc1 = __builtin_amdgcn_wmma_f32_16x16x32_bf16(false, Cf, false, Bf,
                                                   (short)0, acc1, false, false);
  }

  // C layout: VGPR r -> row m = r + 8*half, col = lane%16 within subtile
  if (full) {
#pragma unroll
    for (int r = 0; r < 8; ++r) {
      int node = node0 + r + half * 8;
      hlin[(size_t)node * DD + outCol] = acc0[r];
      hlin[(size_t)(node + 16) * DD + outCol] = acc1[r];
    }
  } else {
#pragma unroll
    for (int r = 0; r < 8; ++r) {
      int node = node0 + r + half * 8;
      if (node < nNodes)      hlin[(size_t)node * DD + outCol] = acc0[r];
      if (node + 16 < nNodes) hlin[(size_t)(node + 16) * DD + outCol] = acc1[r];
    }
  }
}

// ---------------- attention scores per node ----------------
__global__ __launch_bounds__(128) void att_scores(const float* __restrict__ hlin,
                                                  const float* __restrict__ att_s,
                                                  const float* __restrict__ att_d,
                                                  float* __restrict__ asrc,
                                                  float* __restrict__ adst) {
  int node = blockIdx.x;
  int t = threadIdx.x;                             // 0..127, head = t/32
  float h = hlin[(size_t)node * DD + t];
  __shared__ float rs[128], rd[128];
  rs[t] = h * att_s[t];
  rd[t] = h * att_d[t];
  __syncthreads();
  for (int st = 16; st > 0; st >>= 1) {
    if ((t & 31) < st) { rs[t] += rs[t + st]; rd[t] += rd[t + st]; }
    __syncthreads();
  }
  if ((t & 31) == 0) {
    asrc[node * HH + (t >> 5)] = rs[t];
    adst[node * HH + (t >> 5)] = rd[t];
  }
}

// ---------------- edge passes ----------------
__global__ void edge_pass1(const int* __restrict__ ei, const float* __restrict__ asrc,
                           const float* __restrict__ adst, float* __restrict__ ebuf,
                           unsigned* __restrict__ mbuf) {
  int t = blockIdx.x * blockDim.x + threadIdx.x;
  if (t >= ETOT * HH) return;
  int e = t >> 2, h = t & 3;
  int src, dst;
  if (e < EE) { src = ei[e]; dst = ei[EE + e]; } else { src = dst = e - EE; }
  float v = asrc[src * HH + h] + adst[dst * HH + h];
  v = v > 0.f ? v : 0.2f * v;                      // leaky_relu(0.2)
  ebuf[t] = v;
  atomicMax(&mbuf[dst * HH + h], fkey(v));
}

__global__ void edge_pass2(const int* __restrict__ ei, float* __restrict__ ebuf,
                           const unsigned* __restrict__ mbuf, float* __restrict__ sbuf) {
  int t = blockIdx.x * blockDim.x + threadIdx.x;
  if (t >= ETOT * HH) return;
  int e = t >> 2, h = t & 3;
  int dst = (e < EE) ? ei[EE + e] : (e - EE);
  float ex = __expf(ebuf[t] - funkey(mbuf[dst * HH + h]));
  ebuf[t] = ex;
  atomicAdd(&sbuf[dst * HH + h], ex);
}

__global__ void edge_pass3(const int* __restrict__ ei, const float* __restrict__ ebuf,
                           const float* __restrict__ sbuf, const float* __restrict__ hlin,
                           float* __restrict__ agg) {
  long gid = (long)blockIdx.x * blockDim.x + threadIdx.x;
  int e = (int)(gid >> 5), lane = (int)(gid & 31);
  if (e >= ETOT) return;
  int src, dst;
  if (e < EE) { src = ei[e]; dst = ei[EE + e]; } else { src = dst = e - EE; }
#pragma unroll
  for (int h = 0; h < HH; ++h) {                   // ch = lane + 32*h, head = h
    int ch = lane + 32 * h;
    float alpha = ebuf[e * HH + h] / (sbuf[dst * HH + h] + 1e-16f);
    atomicAdd(&agg[(size_t)dst * DD + ch], hlin[(size_t)src * DD + ch] * alpha);
  }
}

__global__ void finalize_layer(const float* __restrict__ agg, const float* __restrict__ bias,
                               float* __restrict__ hbuf) {
  long t = (long)blockIdx.x * blockDim.x + threadIdx.x;
  if (t >= (long)NN * DD) return;
  float v = agg[t] + bias[(int)(t & (DD - 1))];
  hbuf[t] = siluf(v);
}

__global__ void pool_kernel(const float* __restrict__ hbuf, const int* __restrict__ batch,
                            float* __restrict__ pooled) {
  long t = (long)blockIdx.x * blockDim.x + threadIdx.x;
  if (t >= (long)NN * DD) return;
  int n = (int)(t >> 7), ch = (int)(t & (DD - 1));
  atomicAdd(&pooled[(size_t)batch[n] * DD + ch], hbuf[t]);
}

// ---------------- readout KANLinear [128 -> 10] + log_softmax (f32) ----------------
__global__ __launch_bounds__(128) void readout(const float* __restrict__ pooled,
                                               const float* __restrict__ rbw,
                                               const float* __restrict__ rsw,
                                               const float* __restrict__ rsc,
                                               float* __restrict__ out) {
  int g = blockIdx.x, t = threadIdx.x;             // t = input channel
  float x = pooled[(size_t)g * DD + t];
  float s = siluf(x);
  float bs[7]; bspline7(x, bs);
  __shared__ float red[128];
  __shared__ float logits[NCC];
  for (int o = 0; o < NCC; ++o) {
    int wi = o * DD + t;
    float p = s * rbw[wi];
    float sc = rsc[wi];
#pragma unroll
    for (int b = 0; b < 7; ++b) p += bs[b] * rsw[(size_t)wi * NBB + b] * sc;
    red[t] = p;
    __syncthreads();
    for (int st = 64; st > 0; st >>= 1) {
      if (t < st) red[t] += red[t + st];
      __syncthreads();
    }
    if (t == 0) logits[o] = red[0];
    __syncthreads();
  }
  if (t == 0) {
    float mx = logits[0];
    for (int o = 1; o < NCC; ++o) mx = fmaxf(mx, logits[o]);
    float se = 0.f;
    for (int o = 0; o < NCC; ++o) se += expf(logits[o] - mx);
    float lse = logf(se);
    for (int o = 0; o < NCC; ++o) out[g * NCC + o] = logits[o] - mx - lse;
  }
}

// ---------------- host launcher ----------------
static inline long cdivl(long a, long b) { return (a + b - 1) / b; }

extern "C" void kernel_launch(void* const* d_in, const int* in_sizes, int n_in,
                              void* d_out, int out_size, void* d_ws, size_t ws_size,
                              hipStream_t stream) {
  const float* x        = (const float*)d_in[0];
  const int*   ei       = (const int*)d_in[1];
  const int*   batch    = (const int*)d_in[2];
  const float* base_w   = (const float*)d_in[3];
  const float* spline_w = (const float*)d_in[4];
  const float* scaler   = (const float*)d_in[5];
  const float* att_src  = (const float*)d_in[6];
  const float* att_dst  = (const float*)d_in[7];
  const float* bias     = (const float*)d_in[8];
  const float* ro_bw    = (const float*)d_in[9];
  const float* ro_sw    = (const float*)d_in[10];
  const float* ro_sc    = (const float*)d_in[11];
  float* out = (float*)d_out;

  // workspace carve-up (256B aligned)
  char* base = (char*)d_ws;
  size_t off = 0;
  auto carve = [&](size_t bytes) { char* p = base + off; off = (off + bytes + 255) & ~(size_t)255; return p; };
  float*          hbuf   = (float*)carve((size_t)NN * DD * 4);
  float*          hlin   = (float*)carve((size_t)NN * DD * 4);
  float*          agg    = (float*)carve((size_t)NN * DD * 4);
  float*          asrc   = (float*)carve((size_t)NN * HH * 4);
  float*          adst   = (float*)carve((size_t)NN * HH * 4);
  unsigned*       mbuf   = (unsigned*)carve((size_t)NN * HH * 4);
  float*          sbuf   = (float*)carve((size_t)NN * HH * 4);
  float*          ebuf   = (float*)carve((size_t)ETOT * HH * 4);
  unsigned short* wpack  = (unsigned short*)carve((size_t)LL * DD * KPACK * 2);
  float*          pooled = (float*)carve((size_t)NGG * DD * 4);
  (void)ws_size; (void)n_in; (void)in_sizes; (void)out_size;

  const int TB = 256;

  pack_weights<<<cdivl(LL * DD * DD, TB), TB, 0, stream>>>(base_w, spline_w, scaler, wpack);

  for (int l = 0; l < LL; ++l) {
    zero_b32<<<cdivl((long)NN * DD, TB), TB, 0, stream>>>((unsigned*)agg, (long)NN * DD);
    zero_b32<<<cdivl((long)NN * HH, TB), TB, 0, stream>>>(mbuf, (long)NN * HH);
    zero_b32<<<cdivl((long)NN * HH, TB), TB, 0, stream>>>((unsigned*)sbuf, (long)NN * HH);

    const float* xin = (l == 0) ? x : hbuf;
    kan_gemm<<<cdivl(NN, MT), TB, 0, stream>>>(xin, wpack + (size_t)l * DD * KPACK, hlin, NN);
    att_scores<<<NN, 128, 0, stream>>>(hlin, att_src + (size_t)l * DD, att_dst + (size_t)l * DD,
                                       asrc, adst);
    edge_pass1<<<cdivl((long)ETOT * HH, TB), TB, 0, stream>>>(ei, asrc, adst, ebuf, mbuf);
    edge_pass2<<<cdivl((long)ETOT * HH, TB), TB, 0, stream>>>(ei, ebuf, mbuf, sbuf);
    edge_pass3<<<cdivl((long)ETOT * 32, TB), TB, 0, stream>>>(ei, ebuf, sbuf, hlin, agg);
    finalize_layer<<<cdivl((long)NN * DD, TB), TB, 0, stream>>>(agg, bias + (size_t)l * DD, hbuf);
  }

  zero_b32<<<cdivl((long)NGG * DD, TB), TB, 0, stream>>>((unsigned*)pooled, (long)NGG * DD);
  pool_kernel<<<cdivl((long)NN * DD, TB), TB, 0, stream>>>(hbuf, batch, pooled);
  readout<<<NGG, 128, 0, stream>>>(pooled, ro_bw, ro_sw, ro_sc, out);
}
